// PointNet2SemanticSegmentation_36026185679271
// MI455X (gfx1250) — matrix operand order repaired and checked by reading, not compiled, not run on hardware
//
#include <hip/hip_runtime.h>

// ---------------------------------------------------------------------------
// PointNet++ semantic segmentation for MI455X (gfx1250, wave32, WMMA)
// ---------------------------------------------------------------------------

typedef __attribute__((ext_vector_type(16))) _Float16 v16h;
typedef __attribute__((ext_vector_type(8)))  _Float16 v8h;
typedef __attribute__((ext_vector_type(8)))  float    v8f;

#define BB 4
#define NN 8192
#define BN_EPS 1e-5f

static __host__ __device__ inline int pad16(int x) { return (x + 15) & ~15; }
static __host__ __device__ inline int pad32(int x) { return (x + 31) & ~31; }

// ---------------------------------------------------------------------------
// split points[B,N,4] -> xyz[B,N,3], feat[B,N,1]
// ---------------------------------------------------------------------------
__global__ void split_points_kernel(const float* __restrict__ pts,
                                    float* __restrict__ xyz,
                                    float* __restrict__ feat, int total) {
  int i = blockIdx.x * blockDim.x + threadIdx.x;
  if (i >= total) return;
  xyz[i * 3 + 0] = pts[i * 4 + 0];
  xyz[i * 3 + 1] = pts[i * 4 + 1];
  xyz[i * 3 + 2] = pts[i * 4 + 2];
  feat[i] = pts[i * 4 + 3];
}

// ---------------------------------------------------------------------------
// Farthest point sampling: one workgroup per batch, dist[] lives in LDS.
// ---------------------------------------------------------------------------
__global__ __launch_bounds__(1024)
void fps_kernel(const float* __restrict__ xyz, int* __restrict__ out,
                int N, int npoint) {
  const int b = blockIdx.x;
  const float* p = xyz + (size_t)b * N * 3;
  __shared__ float dist[NN];   // 32 KB, << 320 KB/WGP
  __shared__ float rv[1024];
  __shared__ int   ri[1024];
  __shared__ int   curs;
  const int tid = threadIdx.x;
  for (int i = tid; i < N; i += blockDim.x) dist[i] = 1e10f;
  if (tid == 0) curs = 0;
  __syncthreads();
  for (int it = 0; it < npoint; ++it) {
    const int cur = curs;
    if (tid == 0) out[b * npoint + it] = cur;
    const float cx = p[cur * 3 + 0], cy = p[cur * 3 + 1], cz = p[cur * 3 + 2];
    float bestv = -1.0f; int besti = 0;
    for (int i = tid; i < N; i += blockDim.x) {
      float dx = p[i * 3 + 0] - cx, dy = p[i * 3 + 1] - cy, dz = p[i * 3 + 2] - cz;
      float d = dx * dx + dy * dy + dz * dz;
      float dm = fminf(dist[i], d);
      dist[i] = dm;
      if (dm > bestv) { bestv = dm; besti = i; }
    }
    rv[tid] = bestv; ri[tid] = besti;
    __syncthreads();
    for (int off = 512; off > 0; off >>= 1) {
      if (tid < off) {
        if (rv[tid + off] > rv[tid] ||
            (rv[tid + off] == rv[tid] && ri[tid + off] < ri[tid])) {
          rv[tid] = rv[tid + off]; ri[tid] = ri[tid + off];
        }
      }
      __syncthreads();
    }
    if (tid == 0) curs = ri[0];
    __syncthreads();
  }
}

// ---------------------------------------------------------------------------
__global__ void gather_xyz_kernel(const float* __restrict__ xyz,
                                  const int* __restrict__ idx,
                                  float* __restrict__ out,
                                  int N, int npoint, int total) {
  int i = blockIdx.x * blockDim.x + threadIdx.x;
  if (i >= total) return;
  int b = i / npoint;
  int id = idx[i];
  const float* s = xyz + ((size_t)b * N + id) * 3;
  out[i * 3 + 0] = s[0];
  out[i * 3 + 1] = s[1];
  out[i * 3 + 2] = s[2];
}

// ---------------------------------------------------------------------------
// Ball query: first `nsample` points (in index order) within radius.
// ---------------------------------------------------------------------------
__global__ void query_ball_kernel(const float* __restrict__ xyz,
                                  const float* __restrict__ nxyz,
                                  int* __restrict__ idx,
                                  int N, int npoint, int total,
                                  float r2, int nsample) {
  int i = blockIdx.x * blockDim.x + threadIdx.x;
  if (i >= total) return;
  int b = i / npoint;
  const float* p = xyz + (size_t)b * N * 3;
  const float cx = nxyz[i * 3 + 0], cy = nxyz[i * 3 + 1], cz = nxyz[i * 3 + 2];
  int* o = idx + (size_t)i * nsample;
  int cnt = 0;
  for (int j = 0; j < N && cnt < nsample; ++j) {
    __builtin_prefetch(p + (j + 96) * 3, 0, 0);  // global_prefetch_b8
    float dx = p[j * 3 + 0] - cx, dy = p[j * 3 + 1] - cy, dz = p[j * 3 + 2] - cz;
    if (dx * dx + dy * dy + dz * dz <= r2) o[cnt++] = j;
  }
  int f = (cnt > 0) ? o[0] : 0;
  for (int k = cnt; k < nsample; ++k) o[k] = f;
}

// ---------------------------------------------------------------------------
// Build grouped GEMM input X[B*np*ns, Kp] fp16 = [xyz[idx]-new_xyz | feats[idx] | 0pad]
// ---------------------------------------------------------------------------
__global__ void group_concat_kernel(const float* __restrict__ xyz,
                                    const float* __restrict__ feats,
                                    const float* __restrict__ nxyz,
                                    const int* __restrict__ idx,
                                    _Float16* __restrict__ X,
                                    int N, int npoint, int nsample, int Cin,
                                    int Kp, int total) {
  int i = blockIdx.x * blockDim.x + threadIdx.x;
  if (i >= total) return;
  int c = i % Kp;
  int r = i / Kp;              // row: bp*nsample + s
  int bp = r / nsample;        // b*npoint + p
  int b = bp / npoint;
  int id = idx[r];
  float v = 0.0f;
  if (c < 3)
    v = xyz[((size_t)b * N + id) * 3 + c] - nxyz[(size_t)bp * 3 + c];
  else if (c < 3 + Cin)
    v = feats[((size_t)b * N + id) * Cin + (c - 3)];
  X[i] = (_Float16)v;
}

// ---------------------------------------------------------------------------
// Pack fp32 weight [Nout,K] -> fp16 [Npad,Kp] (zero padded)
// ---------------------------------------------------------------------------
__global__ void pack_w_kernel(const float* __restrict__ W,
                              _Float16* __restrict__ Wh,
                              int Nout, int K, int Npad, int Kp) {
  int i = blockIdx.x * blockDim.x + threadIdx.x;
  if (i >= Npad * Kp) return;
  int n = i / Kp, k = i % Kp;
  float v = (n < Nout && k < K) ? W[(size_t)n * K + k] : 0.0f;
  Wh[i] = (_Float16)v;
}

// ---------------------------------------------------------------------------
// WMMA GEMM: Y[M,Nld] = X[M,Kp](f16) * Wh[Npad,Kp]^T (f16) + bias, f32 out.
// One wave per 16x16 tile; K looped in steps of 32 via v_wmma_f32_16x16x32_f16.
// ---------------------------------------------------------------------------
__global__ __launch_bounds__(128)
void gemm_wmma_kernel(const _Float16* __restrict__ X,
                      const _Float16* __restrict__ Wh,
                      const float* __restrict__ bias,
                      float* __restrict__ Y,
                      int M, int Kp, int Nout, int tilesN, int Nld) {
  int wave = (int)((blockIdx.x * blockDim.x + threadIdx.x) >> 5);
  int lane = threadIdx.x & 31;
  int tm = wave / tilesN;
  if (tm * 16 >= M) return;                       // wave-uniform, EXEC stays full
  int tn = wave % tilesN;
  int hi = (lane >= 16);
  // A fragment: row M = tm*16 + (lane&15); K = kbase+{0..7} and kbase+16+{0..7}
  const _Float16* xrow = X + (size_t)(tm * 16 + (lane & 15)) * Kp + (hi ? 8 : 0);
  // B fragment: col N = tn*16 + (lane&15); K = kbase+{0..7} and kbase+16+{0..7}
  const _Float16* wrow = Wh + (size_t)(tn * 16 + (lane & 15)) * Kp + (hi ? 8 : 0);
  v8f acc = {};
  for (int k0 = 0; k0 < Kp; k0 += 32) {
    v8h alo = *(const v8h*)(xrow + k0);
    v8h ahi = *(const v8h*)(xrow + k0 + 16);
    v8h blo = *(const v8h*)(wrow + k0);
    v8h bhi = *(const v8h*)(wrow + k0 + 16);
    v16h b;
    v16h a;
#pragma unroll
    for (int e = 0; e < 8; ++e) { a[e] = alo[e]; a[e + 8] = ahi[e]; b[e] = blo[e]; b[e + 8] = bhi[e]; }
    acc = __builtin_amdgcn_wmma_f32_16x16x32_f16(
        false, a, false, b, (short)0, acc, false, false);
  }
  int n = tn * 16 + (lane & 15);
  if (n < Nout) {
    float bv = bias[n];
    int mbase = tm * 16 + (hi ? 8 : 0);
#pragma unroll
    for (int v = 0; v < 8; ++v)
      Y[(size_t)(mbase + v) * Nld + n] = acc[v] + bv;
  }
}

// ---------------------------------------------------------------------------
// Per-channel sum / sumsq: one block per channel (no atomics, no pre-zero).
// ---------------------------------------------------------------------------
__global__ __launch_bounds__(256)
void bn_stats_kernel(const float* __restrict__ Y, float* __restrict__ stats,
                     int M, int Nld) {
  int n = blockIdx.x;
  float s = 0.f, ss = 0.f;
  for (int m = threadIdx.x; m < M; m += blockDim.x) {
    float v = Y[(size_t)m * Nld + n];
    s += v; ss += v * v;
  }
  __shared__ float sh[512];
  sh[threadIdx.x] = s; sh[threadIdx.x + 256] = ss;
  __syncthreads();
  for (int off = 128; off > 0; off >>= 1) {
    if (threadIdx.x < off) {
      sh[threadIdx.x] += sh[threadIdx.x + off];
      sh[threadIdx.x + 256] += sh[threadIdx.x + 256 + off];
    }
    __syncthreads();
  }
  if (threadIdx.x == 0) { stats[n * 2] = sh[0]; stats[n * 2 + 1] = sh[256]; }
}

// ---------------------------------------------------------------------------
// BN + ReLU; writes next-layer fp16 X (incl. K-pad zeros) and optional fp32.
// ---------------------------------------------------------------------------
__global__ void bn_relu_kernel(const float* __restrict__ Y,
                               const float* __restrict__ g,
                               const float* __restrict__ be,
                               const float* __restrict__ stats,
                               _Float16* __restrict__ Xout,
                               float* __restrict__ Fout,
                               int M, int Nld, int Nout, int KpNext, int total) {
  int i = blockIdx.x * blockDim.x + threadIdx.x;
  if (i >= total) return;
  int m = i / KpNext, n = i % KpNext;
  float r = 0.0f;
  if (n < Nout) {
    float mean = stats[n * 2] / (float)M;
    float var = stats[n * 2 + 1] / (float)M - mean * mean;
    float x = Y[(size_t)m * Nld + n];
    r = fmaxf(g[n] * (x - mean) * rsqrtf(var + BN_EPS) + be[n], 0.0f);
    if (Fout) Fout[(size_t)m * Nout + n] = r;
  }
  Xout[i] = (_Float16)r;
}

// ---------------------------------------------------------------------------
// Max over nsample (reads post-BN fp16 X), fp32 stage features out.
// ---------------------------------------------------------------------------
__global__ void maxpool_kernel(const _Float16* __restrict__ X,
                               float* __restrict__ out,
                               int nsample, int C, int Kp, int total) {
  int i = blockIdx.x * blockDim.x + threadIdx.x;
  if (i >= total) return;
  int c = i % C;
  int bp = i / C;
  const _Float16* x = X + (size_t)bp * nsample * Kp + c;
  float m = -1e30f;
  for (int s = 0; s < nsample; ++s) m = fmaxf(m, (float)x[(size_t)s * Kp]);
  out[i] = m;
}

// ---------------------------------------------------------------------------
// 3-NN (brute force over N2 centroids) with inverse-distance weights.
// ---------------------------------------------------------------------------
__global__ void knn3_kernel(const float* __restrict__ xyz1,
                            const float* __restrict__ xyz2,
                            int* __restrict__ idx3, float* __restrict__ w3,
                            int N1, int N2, int total) {
  int i = blockIdx.x * blockDim.x + threadIdx.x;
  if (i >= total) return;
  int b = i / N1;
  const float qx = xyz1[i * 3 + 0], qy = xyz1[i * 3 + 1], qz = xyz1[i * 3 + 2];
  const float* p = xyz2 + (size_t)b * N2 * 3;
  float d0 = 1e30f, d1 = 1e30f, d2 = 1e30f;
  int i0 = 0, i1 = 0, i2 = 0;
  for (int j = 0; j < N2; ++j) {
    float dx = p[j * 3 + 0] - qx, dy = p[j * 3 + 1] - qy, dz = p[j * 3 + 2] - qz;
    float d = dx * dx + dy * dy + dz * dz;
    if (d < d0)      { d2 = d1; i2 = i1; d1 = d0; i1 = i0; d0 = d; i0 = j; }
    else if (d < d1) { d2 = d1; i2 = i1; d1 = d; i1 = j; }
    else if (d < d2) { d2 = d; i2 = j; }
  }
  float w0 = 1.f / (d0 + 1e-8f), w1 = 1.f / (d1 + 1e-8f), w2 = 1.f / (d2 + 1e-8f);
  float ws = w0 + w1 + w2;
  idx3[i * 3 + 0] = i0; idx3[i * 3 + 1] = i1; idx3[i * 3 + 2] = i2;
  w3[i * 3 + 0] = w0 / ws; w3[i * 3 + 1] = w1 / ws; w3[i * 3 + 2] = w2 / ws;
}

// ---------------------------------------------------------------------------
// X[B*N1, Kp] fp16 = [p1 | sum_j w_j * p2[idx_j] | 0pad]
// ---------------------------------------------------------------------------
__global__ void interp_concat_kernel(const float* __restrict__ p1, int C1,
                                     const float* __restrict__ p2, int C2,
                                     const int* __restrict__ idx3,
                                     const float* __restrict__ w3,
                                     _Float16* __restrict__ X,
                                     int N1, int N2, int Kp, int total) {
  int i = blockIdx.x * blockDim.x + threadIdx.x;
  if (i >= total) return;
  int c = i % Kp;
  int r = i / Kp;              // b*N1 + n
  int b = r / N1;
  float v = 0.0f;
  if (c < C1) {
    v = p1[(size_t)r * C1 + c];
  } else if (c < C1 + C2) {
    int cc = c - C1;
    size_t base = (size_t)b * N2 * C2;
    v = w3[r * 3 + 0] * p2[base + (size_t)idx3[r * 3 + 0] * C2 + cc]
      + w3[r * 3 + 1] * p2[base + (size_t)idx3[r * 3 + 1] * C2 + cc]
      + w3[r * 3 + 2] * p2[base + (size_t)idx3[r * 3 + 2] * C2 + cc];
  }
  X[i] = (_Float16)v;
}

// ===========================================================================
// Host-side helpers
// ===========================================================================
static void run_layer(const float* W, const float* bias, const float* g,
                      const float* be, const _Float16* Xin, _Float16* Xout,
                      float* Fout, _Float16* Wh, float* Y, float* stats,
                      int M, int K, int Kp, int Nout, hipStream_t stream) {
  int Npad = pad16(Nout);
  int tilesN = Npad / 16;
  int wtot = Npad * Kp;
  pack_w_kernel<<<(wtot + 255) / 256, 256, 0, stream>>>(W, Wh, Nout, K, Npad, Kp);
  int waves = (M / 16) * tilesN;
  gemm_wmma_kernel<<<(waves + 3) / 4, 128, 0, stream>>>(Xin, Wh, bias, Y, M, Kp,
                                                        Nout, tilesN, Nout);
  bn_stats_kernel<<<Nout, 256, 0, stream>>>(Y, stats, M, Nout);
  int KpN = pad32(Nout);
  int tot = M * KpN;
  bn_relu_kernel<<<(tot + 255) / 256, 256, 0, stream>>>(Y, g, be, stats, Xout,
                                                        Fout, M, Nout, Nout, KpN, tot);
}

static void run_sa(const float* xyz_in, const float* feats_in, int Nin, int Cin,
                   int npoint, float radius, void* const* d_in, int pb,
                   const int* mlp, int nl, float* new_xyz, float* feats_out,
                   int* fpsidx, int* ballidx, _Float16* X, _Float16* Wh,
                   float* Y, float* stats, hipStream_t stream) {
  const int nsample = 32;
  fps_kernel<<<BB, 1024, 0, stream>>>(xyz_in, fpsidx, Nin, npoint);
  int tot1 = BB * npoint;
  gather_xyz_kernel<<<(tot1 + 255) / 256, 256, 0, stream>>>(xyz_in, fpsidx,
                                                            new_xyz, Nin, npoint, tot1);
  query_ball_kernel<<<(tot1 + 255) / 256, 256, 0, stream>>>(
      xyz_in, new_xyz, ballidx, Nin, npoint, tot1, radius * radius, nsample);
  int K = 3 + Cin;
  int Kp = pad32(K);
  int M = BB * npoint * nsample;
  int tg = M * Kp;
  group_concat_kernel<<<(tg + 255) / 256, 256, 0, stream>>>(
      xyz_in, feats_in, new_xyz, ballidx, X, Nin, npoint, nsample, Cin, Kp, tg);
  for (int i = 0; i < nl; ++i) {
    int Nout = mlp[i];
    run_layer((const float*)d_in[pb + 4 * i + 0], (const float*)d_in[pb + 4 * i + 1],
              (const float*)d_in[pb + 4 * i + 2], (const float*)d_in[pb + 4 * i + 3],
              X, X, nullptr, Wh, Y, stats, M, K, Kp, Nout, stream);
    K = Nout; Kp = pad32(Nout);
  }
  int Clast = mlp[nl - 1];
  int tm = BB * npoint * Clast;
  maxpool_kernel<<<(tm + 255) / 256, 256, 0, stream>>>(X, feats_out, nsample,
                                                       Clast, Kp, tm);
}

static void run_fp(const float* xyz1, int N1, const float* xyz2, int N2,
                   const float* p1, int C1, const float* p2, int C2,
                   void* const* d_in, int pb, const int* mlp, int nl,
                   float* feats_out, int* knnidx, float* knnw, _Float16* X,
                   _Float16* Wh, float* Y, float* stats, hipStream_t stream) {
  int tot = BB * N1;
  knn3_kernel<<<(tot + 255) / 256, 256, 0, stream>>>(xyz1, xyz2, knnidx, knnw,
                                                     N1, N2, tot);
  int K = C1 + C2;
  int Kp = pad32(K);
  int tg = tot * Kp;
  interp_concat_kernel<<<(tg + 255) / 256, 256, 0, stream>>>(
      p1, C1, p2, C2, knnidx, knnw, X, N1, N2, Kp, tg);
  int M = tot;
  for (int i = 0; i < nl; ++i) {
    int Nout = mlp[i];
    float* Fout = (i == nl - 1) ? feats_out : nullptr;
    run_layer((const float*)d_in[pb + 4 * i + 0], (const float*)d_in[pb + 4 * i + 1],
              (const float*)d_in[pb + 4 * i + 2], (const float*)d_in[pb + 4 * i + 3],
              X, X, Fout, Wh, Y, stats, M, K, Kp, Nout, stream);
    K = Nout; Kp = pad32(Nout);
  }
}

// ===========================================================================
extern "C" void kernel_launch(void* const* d_in, const int* in_sizes, int n_in,
                              void* d_out, int out_size, void* d_ws, size_t ws_size,
                              hipStream_t stream) {
  (void)in_sizes; (void)n_in; (void)out_size;
  const float* points = (const float*)d_in[0];

  char* base = (char*)d_ws;
  char* ws = base;
  auto alloc = [&](size_t bytes) -> void* {
    void* p = (void*)ws;
    ws += (bytes + 255) & ~((size_t)255);
    return p;
  };

  float* l0x = (float*)alloc((size_t)BB * NN * 3 * 4);
  float* l0p = (float*)alloc((size_t)BB * NN * 1 * 4);
  float* l1x = (float*)alloc((size_t)BB * 1024 * 3 * 4);
  float* l1p = (float*)alloc((size_t)BB * 1024 * 64 * 4);
  float* l2x = (float*)alloc((size_t)BB * 256 * 3 * 4);
  float* l2p = (float*)alloc((size_t)BB * 256 * 128 * 4);
  float* l3x = (float*)alloc((size_t)BB * 64 * 3 * 4);
  float* l3p = (float*)alloc((size_t)BB * 64 * 256 * 4);
  float* l4x = (float*)alloc((size_t)BB * 16 * 3 * 4);
  float* l4p = (float*)alloc((size_t)BB * 16 * 512 * 4);
  float* l3q = (float*)alloc((size_t)BB * 64 * 256 * 4);
  float* l2q = (float*)alloc((size_t)BB * 256 * 256 * 4);
  float* l1q = (float*)alloc((size_t)BB * 1024 * 128 * 4);
  int* fpsidx = (int*)alloc((size_t)BB * 1024 * 4);
  int* ballidx = (int*)alloc((size_t)BB * 1024 * 32 * 4);
  int* knnidx = (int*)alloc((size_t)BB * NN * 3 * 4);
  float* knnw = (float*)alloc((size_t)BB * NN * 3 * 4);
  float* stats = (float*)alloc(512 * 2 * 4);
  _Float16* Wh = (_Float16*)alloc((size_t)1 << 20);
  _Float16* X = (_Float16*)alloc((size_t)131072 * 64 * 2);
  float* Y = (float*)alloc((size_t)131072 * 64 * 4);
  if ((size_t)(ws - base) > ws_size) return;  // workspace too small: bail out

  int tot0 = BB * NN;
  split_points_kernel<<<(tot0 + 255) / 256, 256, 0, stream>>>(points, l0x, l0p, tot0);

  // ---- Set abstraction stages -------------------------------------------
  { int mlp[3] = {32, 32, 64};
    run_sa(l0x, l0p, NN, 1, 1024, 0.1f, d_in, 1, mlp, 3, l1x, l1p,
           fpsidx, ballidx, X, Wh, Y, stats, stream); }
  { int mlp[3] = {64, 64, 128};
    run_sa(l1x, l1p, 1024, 64, 256, 0.2f, d_in, 13, mlp, 3, l2x, l2p,
           fpsidx, ballidx, X, Wh, Y, stats, stream); }
  { int mlp[3] = {128, 128, 256};
    run_sa(l2x, l2p, 256, 128, 64, 0.4f, d_in, 25, mlp, 3, l3x, l3p,
           fpsidx, ballidx, X, Wh, Y, stats, stream); }
  { int mlp[3] = {256, 256, 512};
    run_sa(l3x, l3p, 64, 256, 16, 0.8f, d_in, 37, mlp, 3, l4x, l4p,
           fpsidx, ballidx, X, Wh, Y, stats, stream); }

  // ---- Feature propagation stages ---------------------------------------
  { int mlp[2] = {256, 256};
    run_fp(l3x, 64, l4x, 16, l3p, 256, l4p, 512, d_in, 49, mlp, 2, l3q,
           knnidx, knnw, X, Wh, Y, stats, stream); }
  { int mlp[2] = {256, 256};
    run_fp(l2x, 256, l3x, 64, l2p, 128, l3q, 256, d_in, 57, mlp, 2, l2q,
           knnidx, knnw, X, Wh, Y, stats, stream); }
  { int mlp[2] = {256, 128};
    run_fp(l1x, 1024, l2x, 256, l1p, 64, l2q, 256, d_in, 65, mlp, 2, l1q,
           knnidx, knnw, X, Wh, Y, stats, stream); }
  { int mlp[3] = {128, 128, 128};
    run_fp(l0x, NN, l1x, 1024, l0p, 1, l1q, 128, d_in, 73, mlp, 3, nullptr,
           knnidx, knnw, X, Wh, Y, stats, stream); }

  // ---- Head: dense 128->128 (BN+ReLU) then 128->20, logits to d_out -----
  run_layer((const float*)d_in[85], (const float*)d_in[86],
            (const float*)d_in[87], (const float*)d_in[88],
            X, X, nullptr, Wh, Y, stats, BB * NN, 128, 128, 128, stream);
  pack_w_kernel<<<(32 * 128 + 255) / 256, 256, 0, stream>>>(
      (const float*)d_in[89], Wh, 20, 128, 32, 128);
  int waves = (BB * NN / 16) * 2;
  gemm_wmma_kernel<<<(waves + 3) / 4, 128, 0, stream>>>(
      X, Wh, (const float*)d_in[90], (float*)d_out, BB * NN, 128, 20, 2, 20);
}
